// RNA_Model_8272107012890
// MI455X (gfx1250) — matrix operand order repaired and checked by reading, not compiled, not run on hardware
//
#include <hip/hip_runtime.h>
#include <hip/hip_bf16.h>
#include <math.h>

// ---------------- types for CDNA5 WMMA ----------------
typedef __attribute__((ext_vector_type(16))) __bf16 v16bf;
typedef __attribute__((ext_vector_type(8)))  float  v8f;

union BF16x16 {
    uint4  q[2];   // two 128-bit halves
    v16bf  v;
};

__device__ __forceinline__ unsigned short f2bfbits(float f) {
    unsigned u = __float_as_uint(f);
    unsigned r = u + 0x7FFFu + ((u >> 16) & 1u);   // round-to-nearest-even
    return (unsigned short)(r >> 16);
}

__device__ __forceinline__ v8f wmma_bf16(const BF16x16& a, const BF16x16& b, v8f c) {
    return __builtin_amdgcn_wmma_f32_16x16x32_bf16(false, a.v, false, b.v,
                                                   (short)0, c, false, false);
}

// ---------------- model dims ----------------
#define TOK   4096      // B*L
#define Dm    256
#define HD    512       // H*DH
#define NH    8
#define DHh   64
#define SEQL  512
#define NB    8
#define DEPTH 16
#define INNER2 2048     // 2*INNER
#define INNER1 1024     // INNER

// =====================================================================
// weight cast + transpose: W[d][K][N] f32 -> WT[d][N][K] bf16
// =====================================================================
__global__ __launch_bounds__(256) void cast_transpose_kernel(
    const float* __restrict__ W, unsigned short* __restrict__ WT, int K, int N)
{
    size_t idx = (size_t)blockIdx.x * 256 + threadIdx.x;  // over DEPTH*N*K
    size_t perL = (size_t)K * N;
    int d   = (int)(idx / perL);
    size_t rem = idx - (size_t)d * perL;
    int n   = (int)(rem / K);
    int kk  = (int)(rem % K);
    WT[idx] = f2bfbits(W[(size_t)d * perL + (size_t)kk * N + n]);
}

// =====================================================================
// embedding gather + LayerNorm -> f32 residual stream
// =====================================================================
__global__ __launch_bounds__(256) void embed_ln_kernel(
    const int* __restrict__ seq, const float* __restrict__ emb,
    const float* __restrict__ g, const float* __restrict__ b,
    float* __restrict__ x)
{
    __shared__ float red[256];
    int row = blockIdx.x, d = threadIdx.x;
    float v = emb[(size_t)seq[row] * Dm + d];
    red[d] = v; __syncthreads();
    for (int s = 128; s > 0; s >>= 1) { if (d < s) red[d] += red[d + s]; __syncthreads(); }
    float mu = red[0] * (1.f / Dm); __syncthreads();
    float dv = v - mu;
    red[d] = dv * dv; __syncthreads();
    for (int s = 128; s > 0; s >>= 1) { if (d < s) red[d] += red[d + s]; __syncthreads(); }
    float var = red[0] * (1.f / Dm);
    x[(size_t)row * Dm + d] = dv * rsqrtf(var + 1e-5f) * g[d] + b[d];
}

// =====================================================================
// LayerNorm f32 -> bf16 (GEMM A operand)
// =====================================================================
__global__ __launch_bounds__(256) void ln_bf16_kernel(
    const float* __restrict__ x, const float* __restrict__ g,
    const float* __restrict__ b, unsigned short* __restrict__ out)
{
    __shared__ float red[256];
    int row = blockIdx.x, d = threadIdx.x;
    float v = x[(size_t)row * Dm + d];
    red[d] = v; __syncthreads();
    for (int s = 128; s > 0; s >>= 1) { if (d < s) red[d] += red[d + s]; __syncthreads(); }
    float mu = red[0] * (1.f / Dm); __syncthreads();
    float dv = v - mu;
    red[d] = dv * dv; __syncthreads();
    for (int s = 128; s > 0; s >>= 1) { if (d < s) red[d] += red[d + s]; __syncthreads(); }
    float var = red[0] * (1.f / Dm);
    out[(size_t)row * Dm + d] = f2bfbits(dv * rsqrtf(var + 1e-5f) * g[d] + b[d]);
}

// =====================================================================
// Generic bf16 WMMA GEMM: C[MxN] = A[MxK] * WT[NxK]^T
// 256 thr = 8 waves; wave w -> n columns [blockIdx.x*256 + w*32, +32)
// mode 0: store bf16  | mode 1: store f32 + bias | mode 2: f32 += (res+bias)
// =====================================================================
__global__ __launch_bounds__(256) void gemm_bf16_kernel(
    const unsigned short* __restrict__ A, const unsigned short* __restrict__ BT,
    const float* __restrict__ bias, void* __restrict__ C,
    int N, int K, int mode)
{
    int tid  = threadIdx.x;
    int wave = tid >> 5;
    int lane = tid & 31;
    int l15   = lane & 15;
    int khalf = lane >> 4;
    int mt  = blockIdx.y;                       // 16-row tile
    int nb  = blockIdx.x * 256 + wave * 32;     // first output col for this wave
    int row = mt * 16 + l15;

    v8f acc0 = {}; v8f acc1 = {};
    const unsigned short* arow = A + (size_t)row * K;
    const unsigned short* bp0base = BT + (size_t)(nb + l15) * K + khalf * 8;
    const unsigned short* bp1base = BT + (size_t)(nb + 16 + l15) * K + khalf * 8;

    for (int k0 = 0; k0 < K; k0 += 32) {
        BF16x16 a, b0, b1;
        a.q[0]  = *(const uint4*)(arow + k0 + khalf * 8);
        a.q[1]  = *(const uint4*)(arow + k0 + 16 + khalf * 8);
        b0.q[0] = *(const uint4*)(bp0base + k0);
        b0.q[1] = *(const uint4*)(bp0base + k0 + 16);
        b1.q[0] = *(const uint4*)(bp1base + k0);
        b1.q[1] = *(const uint4*)(bp1base + k0 + 16);
        acc0 = wmma_bf16(a, b0, acc0);
        acc1 = wmma_bf16(a, b1, acc1);
    }

    int col0 = nb + l15;
    if (mode == 0) {
        unsigned short* Cb = (unsigned short*)C;
        for (int r = 0; r < 8; ++r) {
            int m = mt * 16 + r + 8 * khalf;
            Cb[(size_t)m * N + col0]      = f2bfbits(acc0[r]);
            Cb[(size_t)m * N + col0 + 16] = f2bfbits(acc1[r]);
        }
    } else if (mode == 1) {
        float* Cf = (float*)C;
        float bb0 = bias ? bias[col0] : 0.f;
        float bb1 = bias ? bias[col0 + 16] : 0.f;
        for (int r = 0; r < 8; ++r) {
            int m = mt * 16 + r + 8 * khalf;
            Cf[(size_t)m * N + col0]      = acc0[r] + bb0;
            Cf[(size_t)m * N + col0 + 16] = acc1[r] + bb1;
        }
    } else {
        float* Cf = (float*)C;
        float bb0 = bias ? bias[col0] : 0.f;
        float bb1 = bias ? bias[col0 + 16] : 0.f;
        for (int r = 0; r < 8; ++r) {
            int m = mt * 16 + r + 8 * khalf;
            Cf[(size_t)m * N + col0]      += acc0[r] + bb0;
            Cf[(size_t)m * N + col0 + 16] += acc1[r] + bb1;
        }
    }
}

// =====================================================================
// v[tok][HD] bf16 -> vT[bh][DH][SEQL] bf16 (so P@V B-frag loads are contiguous)
// =====================================================================
__global__ __launch_bounds__(256) void transpose_v_kernel(
    const unsigned short* __restrict__ v, unsigned short* __restrict__ vT)
{
    size_t idx = (size_t)blockIdx.x * 256 + threadIdx.x;   // over 64*64*512
    int j  = (int)(idx & 511);
    int d  = (int)((idx >> 9) & 63);
    int bh = (int)(idx >> 15);
    int bb = bh >> 3, hh = bh & 7;
    vT[idx] = v[(size_t)(bb * SEQL + j) * HD + hh * DHh + d];
}

// =====================================================================
// Attention: one block per (bh, 16-query tile); WMMA QK^T + softmax + WMMA PV
// =====================================================================
__global__ __launch_bounds__(256) void attn_kernel(
    const unsigned short* __restrict__ q, const unsigned short* __restrict__ k,
    const unsigned short* __restrict__ vT, const int* __restrict__ mask,
    unsigned short* __restrict__ o)
{
    __shared__ float          sS[16][SEQL];     // 32 KB scores
    __shared__ unsigned short sP[16][SEQL];     // 16 KB probs bf16
    __shared__ float          sred[16][16];
    __shared__ float          srow[16];

    int tid = threadIdx.x, wave = tid >> 5, lane = tid & 31;
    int l15 = lane & 15, khalf = lane >> 4;
    int itile = blockIdx.x, bh = blockIdx.y;
    int bb = bh >> 3, hh = bh & 7;
    const float scale = 0.125f;                  // DH^-0.5
    const float slope = exp2f(-(float)(hh + 1)); // 2^(-(h+1)*8/H), H=8

    // ---- Phase 1: scores = q k^T * scale + alibi, masked ----
    {
        const unsigned short* qp =
            q + (size_t)(bb * SEQL + itile * 16 + l15) * HD + hh * DHh;
        BF16x16 a0, a1;
        a0.q[0] = *(const uint4*)(qp + khalf * 8);
        a0.q[1] = *(const uint4*)(qp + 16 + khalf * 8);
        a1.q[0] = *(const uint4*)(qp + 32 + khalf * 8);
        a1.q[1] = *(const uint4*)(qp + 48 + khalf * 8);

        for (int t = 0; t < 4; ++t) {
            int jt = wave * 4 + t;               // 8 waves * 4 = 32 j-tiles
            int jcol = jt * 16 + l15;
            const unsigned short* kp =
                k + (size_t)(bb * SEQL + jcol) * HD + hh * DHh + khalf * 8;
            BF16x16 fb0, fb1;
            fb0.q[0] = *(const uint4*)(kp);
            fb0.q[1] = *(const uint4*)(kp + 16);
            fb1.q[0] = *(const uint4*)(kp + 32);
            fb1.q[1] = *(const uint4*)(kp + 48);
            v8f acc = {};
            acc = wmma_bf16(a0, fb0, acc);
            acc = wmma_bf16(a1, fb1, acc);
            for (int r = 0; r < 8; ++r) {
                int il = r + 8 * khalf;
                int ig = itile * 16 + il;
                int jg = jt * 16 + l15;
                float s = acc[r] * scale + slope * (-fabsf((float)(ig - jg)));
                if (mask[bb * SEQL + jg] == 0) s = -1e9f;
                sS[il][jg] = s;
            }
        }
    }
    __syncthreads();

    // ---- Phase 2: softmax over 512 keys; 16 threads per query row ----
    {
        int row = tid >> 4, sub = tid & 15;
        float m = -3.4e38f;
        for (int j = sub * 32; j < sub * 32 + 32; ++j) m = fmaxf(m, sS[row][j]);
        sred[row][sub] = m; __syncthreads();
        if (sub == 0) {
            float mm = sred[row][0];
            for (int t = 1; t < 16; ++t) mm = fmaxf(mm, sred[row][t]);
            srow[row] = mm;
        }
        __syncthreads();
        float rm = srow[row];
        float sum = 0.f;
        for (int j = sub * 32; j < sub * 32 + 32; ++j) {
            float p = __expf(sS[row][j] - rm);
            sS[row][j] = p; sum += p;
        }
        sred[row][sub] = sum; __syncthreads();
        if (sub == 0) {
            float ss = 0.f;
            for (int t = 0; t < 16; ++t) ss += sred[row][t];
            srow[row] = ss;
        }
        __syncthreads();
        float inv = 1.f / srow[row];
        for (int j = sub * 32; j < sub * 32 + 32; ++j)
            sP[row][j] = f2bfbits(sS[row][j] * inv);
    }
    __syncthreads();

    // ---- Phase 3: O = P @ V, waves 0..3 each own a 16-wide d-tile ----
    if (wave < 4) {
        int dbase = wave * 16;
        v8f acc = {};
        for (int j0 = 0; j0 < SEQL; j0 += 32) {
            BF16x16 a, fb;
            a.q[0] = *(const uint4*)(&sP[l15][j0 + khalf * 8]);
            a.q[1] = *(const uint4*)(&sP[l15][j0 + 16 + khalf * 8]);
            const unsigned short* vp =
                vT + ((size_t)bh * 64 + dbase + l15) * SEQL + j0 + khalf * 8;
            fb.q[0] = *(const uint4*)(vp);
            fb.q[1] = *(const uint4*)(vp + 16);
            acc = wmma_bf16(a, fb, acc);
        }
        for (int r = 0; r < 8; ++r) {
            int il = r + 8 * khalf;
            size_t orow = (size_t)(bb * SEQL + itile * 16 + il);
            o[orow * HD + hh * DHh + dbase + l15] = f2bfbits(acc[r]);
        }
    }
}

// =====================================================================
// GLU gating: g = val * gelu_exact(gate)   (u f32 [TOK][2048] -> bf16 [TOK][1024])
// =====================================================================
__global__ __launch_bounds__(256) void gating_kernel(
    const float* __restrict__ u, unsigned short* __restrict__ g)
{
    size_t idx = (size_t)blockIdx.x * 256 + threadIdx.x;   // over TOK*INNER1
    size_t row = idx >> 10;
    int    c   = (int)(idx & 1023);
    float val  = u[row * INNER2 + c];
    float gate = u[row * INNER2 + INNER1 + c];
    float ge   = 0.5f * gate * (1.f + erff(gate * 0.70710678118654752f));
    g[idx] = f2bfbits(val * ge);
}

// =====================================================================
// Final LN + projection D->2
// =====================================================================
__global__ __launch_bounds__(256) void final_kernel(
    const float* __restrict__ x, const float* __restrict__ g,
    const float* __restrict__ b, const float* __restrict__ Wout,
    const float* __restrict__ bout, float* __restrict__ out)
{
    __shared__ float red[256];
    __shared__ float red2[256];
    int row = blockIdx.x, d = threadIdx.x;
    float v = x[(size_t)row * Dm + d];
    red[d] = v; __syncthreads();
    for (int s = 128; s > 0; s >>= 1) { if (d < s) red[d] += red[d + s]; __syncthreads(); }
    float mu = red[0] * (1.f / Dm); __syncthreads();
    float dv = v - mu;
    red[d] = dv * dv; __syncthreads();
    for (int s = 128; s > 0; s >>= 1) { if (d < s) red[d] += red[d + s]; __syncthreads(); }
    float var = red[0] * (1.f / Dm); __syncthreads();
    float y = dv * rsqrtf(var + 1e-5f) * g[d] + b[d];
    red[d]  = y * Wout[d * 2 + 0];
    red2[d] = y * Wout[d * 2 + 1];
    __syncthreads();
    for (int s = 128; s > 0; s >>= 1) {
        if (d < s) { red[d] += red[d + s]; red2[d] += red2[d + s]; }
        __syncthreads();
    }
    if (d == 0) {
        out[row * 2 + 0] = red[0] + bout[0];
        out[row * 2 + 1] = red2[0] + bout[1];
    }
}

// =====================================================================
// host launcher
// =====================================================================
extern "C" void kernel_launch(void* const* d_in, const int* in_sizes, int n_in,
                              void* d_out, int out_size, void* d_ws, size_t ws_size,
                              hipStream_t stream)
{
    (void)in_sizes; (void)n_in; (void)out_size; (void)ws_size;

    const int*   seq  = (const int*)  d_in[0];
    const int*   mask = (const int*)  d_in[1];
    const float* emb  = (const float*)d_in[2];
    const float* png  = (const float*)d_in[3];
    const float* pnb  = (const float*)d_in[4];
    const float* ln1g = (const float*)d_in[5];
    const float* ln1b = (const float*)d_in[6];
    const float* Wq   = (const float*)d_in[7];
    const float* Wk   = (const float*)d_in[8];
    const float* Wv   = (const float*)d_in[9];
    const float* Wo   = (const float*)d_in[10];
    const float* ln2g = (const float*)d_in[11];
    const float* ln2b = (const float*)d_in[12];
    const float* W1   = (const float*)d_in[13];
    const float* b1   = (const float*)d_in[14];
    const float* W2   = (const float*)d_in[15];
    const float* b2   = (const float*)d_in[16];
    const float* fng  = (const float*)d_in[17];
    const float* fnb  = (const float*)d_in[18];
    const float* Wout = (const float*)d_in[19];
    const float* bout = (const float*)d_in[20];

    char* ws = (char*)d_ws;
    size_t off = 0;
    auto alloc = [&](size_t bytes) -> void* {
        void* p = ws + off;
        off = (off + bytes + 255) & ~(size_t)255;
        return p;
    };

    unsigned short* WqT = (unsigned short*)alloc((size_t)DEPTH * HD * Dm * 2);
    unsigned short* WkT = (unsigned short*)alloc((size_t)DEPTH * HD * Dm * 2);
    unsigned short* WvT = (unsigned short*)alloc((size_t)DEPTH * HD * Dm * 2);
    unsigned short* WoT = (unsigned short*)alloc((size_t)DEPTH * Dm * HD * 2);
    unsigned short* W1T = (unsigned short*)alloc((size_t)DEPTH * INNER2 * Dm * 2);
    unsigned short* W2T = (unsigned short*)alloc((size_t)DEPTH * Dm * INNER1 * 2);
    float*          x    = (float*)         alloc((size_t)TOK * Dm * 4);
    unsigned short* h    = (unsigned short*)alloc((size_t)TOK * Dm * 2);
    unsigned short* qb   = (unsigned short*)alloc((size_t)TOK * HD * 2);
    unsigned short* kb   = (unsigned short*)alloc((size_t)TOK * HD * 2);
    unsigned short* vb   = (unsigned short*)alloc((size_t)TOK * HD * 2);
    unsigned short* vTb  = (unsigned short*)alloc((size_t)TOK * HD * 2);
    unsigned short* ob   = (unsigned short*)alloc((size_t)TOK * HD * 2);
    float*          ub   = (float*)         alloc((size_t)TOK * INNER2 * 4);
    unsigned short* gb   = (unsigned short*)alloc((size_t)TOK * INNER1 * 2);

    // ---- weights -> bf16, transposed to [N][K] per layer ----
    cast_transpose_kernel<<<DEPTH*Dm*HD/256,     256, 0, stream>>>(Wq, WqT, Dm, HD);
    cast_transpose_kernel<<<DEPTH*Dm*HD/256,     256, 0, stream>>>(Wk, WkT, Dm, HD);
    cast_transpose_kernel<<<DEPTH*Dm*HD/256,     256, 0, stream>>>(Wv, WvT, Dm, HD);
    cast_transpose_kernel<<<DEPTH*HD*Dm/256,     256, 0, stream>>>(Wo, WoT, HD, Dm);
    cast_transpose_kernel<<<DEPTH*Dm*INNER2/256, 256, 0, stream>>>(W1, W1T, Dm, INNER2);
    cast_transpose_kernel<<<DEPTH*INNER1*Dm/256, 256, 0, stream>>>(W2, W2T, INNER1, Dm);

    // ---- embed + pre-LN ----
    embed_ln_kernel<<<TOK, 256, 0, stream>>>(seq, emb, png, pnb, x);

    for (int d = 0; d < DEPTH; ++d) {
        // LN1 -> h (bf16)
        ln_bf16_kernel<<<TOK, 256, 0, stream>>>(x, ln1g + d*Dm, ln1b + d*Dm, h);
        // q,k,v = h @ Wq/Wk/Wv   (M=4096, N=512, K=256)
        gemm_bf16_kernel<<<dim3(HD/256, TOK/16), 256, 0, stream>>>(
            h, WqT + (size_t)d*HD*Dm, nullptr, qb, HD, Dm, 0);
        gemm_bf16_kernel<<<dim3(HD/256, TOK/16), 256, 0, stream>>>(
            h, WkT + (size_t)d*HD*Dm, nullptr, kb, HD, Dm, 0);
        gemm_bf16_kernel<<<dim3(HD/256, TOK/16), 256, 0, stream>>>(
            h, WvT + (size_t)d*HD*Dm, nullptr, vb, HD, Dm, 0);
        // v transpose for PV gemm
        transpose_v_kernel<<<TOK*HD/256, 256, 0, stream>>>(vb, vTb);
        // attention
        attn_kernel<<<dim3(SEQL/16, NB*NH), 256, 0, stream>>>(qb, kb, vTb, mask, ob);
        // x += o @ Wo   (M=4096, N=256, K=512)
        gemm_bf16_kernel<<<dim3(Dm/256, TOK/16), 256, 0, stream>>>(
            ob, WoT + (size_t)d*Dm*HD, nullptr, x, Dm, HD, 2);
        // LN2 -> h
        ln_bf16_kernel<<<TOK, 256, 0, stream>>>(x, ln2g + d*Dm, ln2b + d*Dm, h);
        // u = h @ W1 + b1   (M=4096, N=2048, K=256)
        gemm_bf16_kernel<<<dim3(INNER2/256, TOK/16), 256, 0, stream>>>(
            h, W1T + (size_t)d*INNER2*Dm, b1 + (size_t)d*INNER2, ub, INNER2, Dm, 1);
        // g = val * gelu(gate)
        gating_kernel<<<TOK*INNER1/256, 256, 0, stream>>>(ub, gb);
        // x += g @ W2 + b2   (M=4096, N=256, K=1024)
        gemm_bf16_kernel<<<dim3(Dm/256, TOK/16), 256, 0, stream>>>(
            gb, W2T + (size_t)d*Dm*INNER1, b2 + (size_t)d*Dm, x, Dm, INNER1, 2);
    }

    // ---- final LN + projection ----
    final_kernel<<<TOK, 256, 0, stream>>>(x, fng, fnb, Wout, bout, (float*)d_out);
}
